// PatchGraphGATv2_10282151707217
// MI455X (gfx1250) — matrix operand order repaired and checked by reading, not compiled, not run on hardware
//
#include <hip/hip_runtime.h>
#include <hip/hip_bf16.h>

// ---------------------------------------------------------------------------
// GATv2 2-layer forward for MI455X (gfx1250, wave32).
// GEMMs: v_wmma_f32_16x16x32_bf16 (bf16 in, f32 accum).
// Edge phase: wave-per-edge, native f32 global atomics (max/add).
// Requires ~390 MB of d_ws scratch.
// ---------------------------------------------------------------------------

typedef __attribute__((ext_vector_type(16))) __bf16 bf16x16;
typedef __attribute__((ext_vector_type(8)))  __bf16 bf16x8;
typedef __attribute__((ext_vector_type(8)))  float  f32x8;

#define NNODES 50000
#define NEDGES 400000
#define ETOT   (NEDGES + NNODES)   // with self loops
#define INC    256
#define HIDC   128
#define NHEADS 4
#define HC1    (NHEADS * HIDC)     // 512
#define OUTC   256
#define SLOPE  0.2f

// ---------------- small utility kernels ----------------

__global__ void k_fill_f32(float* __restrict__ p, float v, int n) {
  int i = blockIdx.x * blockDim.x + threadIdx.x;
  if (i < n) p[i] = v;
}

__global__ void k_cvt_bf16(const float* __restrict__ s, __bf16* __restrict__ d, int n) {
  int i = blockIdx.x * blockDim.x + threadIdx.x;
  if (i < n) d[i] = (__bf16)s[i];
}

// W [I x O] row-major f32  ->  Wt [O x I] bf16 (transposed)
__global__ void k_transpose_cvt(const float* __restrict__ W, __bf16* __restrict__ Wt,
                                int I, int O) {
  int idx = blockIdx.x * blockDim.x + threadIdx.x;
  if (idx >= I * O) return;
  int o = idx / I, i = idx % I;
  Wt[idx] = (__bf16)W[(long)i * O + o];
}

// ---------------- WMMA GEMM:  C[M x Nc] = A[M x K] @ Bt[Nc x K]^T -----------
// A, Bt bf16; C f32.  M % 16 == 0, Nc % 64 == 0, K % 32 == 0.
// One wave -> 16 rows x 64 cols (4 wmma accumulators); 4 waves per block.
__global__ void __launch_bounds__(128)
k_gemm_bf16_wmma(const __bf16* __restrict__ A, const __bf16* __restrict__ Bt,
                 float* __restrict__ C, int M, int K, int Nc) {
  const int lane = threadIdx.x & 31;
  const int wave = threadIdx.x >> 5;
  const int rowTile = blockIdx.y * 4 + wave;
  if (rowTile * 16 >= M) return;
  const int colBase = blockIdx.x * 64;
  const int m    = lane & 15;     // M index for A, N index for B/C
  const int half = lane >> 4;

  f32x8 acc[4] = {};
  const __bf16* arow = A + (long)(rowTile * 16 + m) * K;

  for (int k0 = 0; k0 < K; k0 += 32) {
    // A tile 16x32: per-lane two contiguous 8-element bf16 chunks (ISA layout)
    union { bf16x16 v; bf16x8 h[2]; } a;
    a.h[0] = *(const bf16x8*)(arow + k0 + half * 8);
    a.h[1] = *(const bf16x8*)(arow + k0 + 16 + half * 8);
    if (k0 + 32 < K) __builtin_prefetch(arow + k0 + 32, 0, 0);   // global_prefetch_b8
#pragma unroll
    for (int j = 0; j < 4; ++j) {
      // B tile 32x16: per-lane two contiguous 8-element bf16 chunks from transposed weights (same k order as A)
      const __bf16* bp = Bt + (long)(colBase + j * 16 + m) * K + k0 + half * 8;
      union { bf16x16 v; bf16x8 h[2]; } bu;
      bu.h[0] = *(const bf16x8*)bp;
      bu.h[1] = *(const bf16x8*)(bp + 16);
      bf16x16 b = bu.v;
      acc[j] = __builtin_amdgcn_wmma_f32_16x16x32_bf16(
          false, a.v, false, b, (short)0, acc[j], false, false);
    }
  }

#pragma unroll
  for (int j = 0; j < 4; ++j) {
#pragma unroll
    for (int v = 0; v < 8; ++v) {
      int row = rowTile * 16 + half * 8 + v;   // C/D layout: lanes16-31 -> M+8
      int col = colBase + j * 16 + m;
      C[(long)row * Nc + col] = acc[j][v];
    }
  }
}

// ---------------- edge kernels (wave32 per edge) ----------------

template <int H, int C>
__global__ void __launch_bounds__(256)
k_edge_scores(const int* __restrict__ esrc, const int* __restrict__ edst,
              const float* __restrict__ xl, const float* __restrict__ xr,
              const float* __restrict__ att,         // [H*C]
              float* __restrict__ ebuf,              // [ETOT*H]
              float* __restrict__ emax) {            // [N*H]
  const int lane = threadIdx.x & 31;
  const int e = (blockIdx.x * blockDim.x + threadIdx.x) >> 5;
  if (e >= ETOT) return;
  const int s = (e < NEDGES) ? esrc[e] : (e - NEDGES);
  const int d = (e < NEDGES) ? edst[e] : (e - NEDGES);
  constexpr int HCn  = H * C;
  constexpr int ITER = HCn / 32;
  constexpr int IPH  = C / 32;      // iterations per head (same head across a wave)
  float hs[H];
#pragma unroll
  for (int h = 0; h < H; ++h) hs[h] = 0.f;
  const float* pl = xl + (long)s * HCn;
  const float* pr = xr + (long)d * HCn;
#pragma unroll
  for (int i = 0; i < ITER; ++i) {
    int cc = lane + 32 * i;
    float g = pl[cc] + pr[cc];
    g = g > 0.f ? g : SLOPE * g;           // leaky_relu
    hs[i / IPH] += g * att[cc];
  }
#pragma unroll
  for (int h = 0; h < H; ++h)
    for (int off = 16; off > 0; off >>= 1)
      hs[h] += __shfl_xor(hs[h], off, 32);
  if (lane == 0) {
#pragma unroll
    for (int h = 0; h < H; ++h) {
      ebuf[(long)e * H + h] = hs[h];
      __hip_atomic_fetch_max(&emax[d * H + h], hs[h],
                             __ATOMIC_RELAXED, __HIP_MEMORY_SCOPE_AGENT);
    }
  }
}

template <int H>
__global__ void k_edge_softmax(const int* __restrict__ edst,
                               float* __restrict__ ebuf,        // in: e, out: p
                               const float* __restrict__ emax,
                               float* __restrict__ denom) {
  int t = blockIdx.x * blockDim.x + threadIdx.x;
  if (t >= ETOT * H) return;
  int e = t / H, h = t % H;
  int d = (e < NEDGES) ? edst[e] : (e - NEDGES);
  float p = __expf(ebuf[t] - emax[d * H + h]);
  ebuf[t] = p;
  __hip_atomic_fetch_add(&denom[d * H + h], p,
                         __ATOMIC_RELAXED, __HIP_MEMORY_SCOPE_AGENT);
}

template <int H, int C>
__global__ void __launch_bounds__(256)
k_edge_aggregate(const int* __restrict__ esrc, const int* __restrict__ edst,
                 const float* __restrict__ xl, const float* __restrict__ ebuf,
                 const float* __restrict__ denom, float* __restrict__ acc) {
  const int lane = threadIdx.x & 31;
  const int e = (blockIdx.x * blockDim.x + threadIdx.x) >> 5;
  if (e >= ETOT) return;
  const int s = (e < NEDGES) ? esrc[e] : (e - NEDGES);
  const int d = (e < NEDGES) ? edst[e] : (e - NEDGES);
  constexpr int HCn = H * C, ITER = HCn / 32, IPH = C / 32;
  float alpha[H];
#pragma unroll
  for (int h = 0; h < H; ++h)
    alpha[h] = ebuf[(long)e * H + h] / denom[d * H + h];
  const float* pl = xl + (long)s * HCn;
  float* pd = acc + (long)d * HCn;
#pragma unroll
  for (int i = 0; i < ITER; ++i) {
    int cc = lane + 32 * i;
    __hip_atomic_fetch_add(&pd[cc], alpha[i / IPH] * pl[cc],
                           __ATOMIC_RELAXED, __HIP_MEMORY_SCOPE_AGENT);
  }
}

// ---------------- finalize kernels ----------------

// layer1: h = relu(acc + b1) -> bf16 for next GEMM
__global__ void k_finalize1(const float* __restrict__ acc, const float* __restrict__ bias,
                            __bf16* __restrict__ hbf, int n) {
  int i = blockIdx.x * blockDim.x + threadIdx.x;
  if (i >= n) return;
  float v = acc[i] + bias[i % HC1];
  hbf[i] = (__bf16)(v > 0.f ? v : 0.f);
}

// layer2: out += b2  (heads=1, mean == identity)
__global__ void k_finalize2(float* __restrict__ out, const float* __restrict__ bias, int n) {
  int i = blockIdx.x * blockDim.x + threadIdx.x;
  if (i >= n) return;
  out[i] = out[i] + bias[i % OUTC];
}

// ---------------------------------------------------------------------------

extern "C" void kernel_launch(void* const* d_in, const int* in_sizes, int n_in,
                              void* d_out, int out_size, void* d_ws, size_t ws_size,
                              hipStream_t stream) {
  const float* x    = (const float*)d_in[0];
  const int*   eidx = (const int*)d_in[1];          // [2, E]
  const float* W1l  = (const float*)d_in[2];
  const float* W1r  = (const float*)d_in[3];
  const float* a1   = (const float*)d_in[4];
  const float* b1   = (const float*)d_in[5];
  const float* W2l  = (const float*)d_in[6];
  const float* W2r  = (const float*)d_in[7];
  const float* a2   = (const float*)d_in[8];
  const float* b2   = (const float*)d_in[9];
  const int* esrc = eidx;
  const int* edst = eidx + NEDGES;
  float* out = (float*)d_out;

  // ---- bump-allocate workspace ----
  char* base = (char*)d_ws;
  size_t off = 0;
  auto alloc = [&](size_t bytes) -> char* {
    char* p = base + off;
    off += (bytes + 255) & ~(size_t)255;
    return p;
  };
  __bf16* xbf   = (__bf16*)alloc((size_t)NNODES * INC * 2);
  __bf16* hbf   = (__bf16*)alloc((size_t)NNODES * HC1 * 2);
  __bf16* wt1l  = (__bf16*)alloc((size_t)INC * HC1 * 2);
  __bf16* wt1r  = (__bf16*)alloc((size_t)INC * HC1 * 2);
  __bf16* wt2l  = (__bf16*)alloc((size_t)HC1 * OUTC * 2);
  __bf16* wt2r  = (__bf16*)alloc((size_t)HC1 * OUTC * 2);
  float*  xlA   = (float*)alloc((size_t)NNODES * HC1 * 4);   // xl1, reused as xl2
  float*  xrA   = (float*)alloc((size_t)NNODES * HC1 * 4);   // xr1, reused as xr2
  float*  acc1  = (float*)alloc((size_t)NNODES * HC1 * 4);
  float*  ebuf  = (float*)alloc((size_t)ETOT * NHEADS * 4);
  float*  emax  = (float*)alloc((size_t)NNODES * NHEADS * 4);
  float*  denom = (float*)alloc((size_t)NNODES * NHEADS * 4);

  const int TPB = 256;
  auto cdiv = [](long a, long b) { return (int)((a + b - 1) / b); };

  // ---- prep: conversions / transposes ----
  k_cvt_bf16<<<cdiv((long)NNODES * INC, TPB), TPB, 0, stream>>>(x, xbf, NNODES * INC);
  k_transpose_cvt<<<cdiv((long)INC * HC1, TPB), TPB, 0, stream>>>(W1l, wt1l, INC, HC1);
  k_transpose_cvt<<<cdiv((long)INC * HC1, TPB), TPB, 0, stream>>>(W1r, wt1r, INC, HC1);
  k_transpose_cvt<<<cdiv((long)HC1 * OUTC, TPB), TPB, 0, stream>>>(W2l, wt2l, HC1, OUTC);
  k_transpose_cvt<<<cdiv((long)HC1 * OUTC, TPB), TPB, 0, stream>>>(W2r, wt2r, HC1, OUTC);

  // ---- layer 1 ----
  k_fill_f32<<<cdiv(NNODES * NHEADS, TPB), TPB, 0, stream>>>(emax, -__builtin_inff(), NNODES * NHEADS);
  k_fill_f32<<<cdiv(NNODES * NHEADS, TPB), TPB, 0, stream>>>(denom, 0.f, NNODES * NHEADS);
  k_fill_f32<<<cdiv((long)NNODES * HC1, TPB), TPB, 0, stream>>>(acc1, 0.f, NNODES * HC1);

  {
    dim3 grid(HC1 / 64, cdiv(NNODES / 16, 4));
    k_gemm_bf16_wmma<<<grid, 128, 0, stream>>>(xbf, wt1l, xlA, NNODES, INC, HC1);
    k_gemm_bf16_wmma<<<grid, 128, 0, stream>>>(xbf, wt1r, xrA, NNODES, INC, HC1);
  }
  k_edge_scores<NHEADS, HIDC><<<cdiv((long)ETOT * 32, TPB), TPB, 0, stream>>>(
      esrc, edst, xlA, xrA, a1, ebuf, emax);
  k_edge_softmax<NHEADS><<<cdiv((long)ETOT * NHEADS, TPB), TPB, 0, stream>>>(
      edst, ebuf, emax, denom);
  k_edge_aggregate<NHEADS, HIDC><<<cdiv((long)ETOT * 32, TPB), TPB, 0, stream>>>(
      esrc, edst, xlA, ebuf, denom, acc1);
  k_finalize1<<<cdiv((long)NNODES * HC1, TPB), TPB, 0, stream>>>(
      acc1, b1, hbf, NNODES * HC1);

  // ---- layer 2 (reuse xlA/xrA/emax/denom/ebuf; accumulate into d_out) ----
  k_fill_f32<<<cdiv(NNODES, TPB), TPB, 0, stream>>>(emax, -__builtin_inff(), NNODES);
  k_fill_f32<<<cdiv(NNODES, TPB), TPB, 0, stream>>>(denom, 0.f, NNODES);
  k_fill_f32<<<cdiv((long)NNODES * OUTC, TPB), TPB, 0, stream>>>(out, 0.f, NNODES * OUTC);

  {
    dim3 grid(OUTC / 64, cdiv(NNODES / 16, 4));
    k_gemm_bf16_wmma<<<grid, 128, 0, stream>>>(hbf, wt2l, xlA, NNODES, HC1, OUTC);
    k_gemm_bf16_wmma<<<grid, 128, 0, stream>>>(hbf, wt2r, xrA, NNODES, HC1, OUTC);
  }
  k_edge_scores<1, OUTC><<<cdiv((long)ETOT * 32, TPB), TPB, 0, stream>>>(
      esrc, edst, xlA, xrA, a2, ebuf, emax);
  k_edge_softmax<1><<<cdiv((long)ETOT, TPB), TPB, 0, stream>>>(
      edst, ebuf, emax, denom);
  k_edge_aggregate<1, OUTC><<<cdiv((long)ETOT * 32, TPB), TPB, 0, stream>>>(
      esrc, edst, xlA, ebuf, denom, out);
  k_finalize2<<<cdiv((long)NNODES * OUTC, TPB), TPB, 0, stream>>>(
      out, b2, NNODES * OUTC);

  (void)in_sizes; (void)n_in; (void)out_size; (void)ws_size;
}
